// GCN_35966056137086
// MI455X (gfx1250) — compile-verified, matrix operand order checked by reading, not run on hardware
//
#include <hip/hip_runtime.h>
#include <hip/hip_bf16.h>
#include <stdint.h>

typedef __attribute__((ext_vector_type(16))) _Float16 v16h;
typedef __attribute__((ext_vector_type(8)))  float    v8f;
typedef __attribute__((ext_vector_type(4)))  int      v4i;

#define N_NODES   100000
#define N_EDGES   640000
#define DIM       128
#define NLAYERS   5
#define NAF       9
#define NBF       3
#define BVOCAB    8
#define LN_EPS    1e-5f

#if defined(__gfx1250__) && __has_builtin(__builtin_amdgcn_global_load_async_to_lds_b128)
#define HAVE_ASYNC_LDS 1
#else
#define HAVE_ASYNC_LDS 0
#endif

typedef __attribute__((address_space(1))) v4i* gv4i_p;   // global int4*
typedef __attribute__((address_space(3))) v4i* lv4i_p;   // LDS int4*

__device__ __forceinline__ void wait_asynccnt0() {
#if __has_builtin(__builtin_amdgcn_s_wait_asynccnt)
    __builtin_amdgcn_s_wait_asynccnt(0);
#elif defined(__gfx1250__)
    asm volatile("s_wait_asynccnt 0" ::: "memory");
#endif
}

// ---------------------------------------------------------------------------
// Atom encoder: h[n,:] = sum_f atom_emb[f, x[n,f], :]; also deg[n] = 1.0
// One wave32 per node, float4 per lane.
// ---------------------------------------------------------------------------
__global__ void __launch_bounds__(256) k_atom_encode(
    const int* __restrict__ x, const float* __restrict__ atom_emb,
    float* __restrict__ h, _Float16* __restrict__ h16, float* __restrict__ deg)
{
    int gid  = blockIdx.x * blockDim.x + threadIdx.x;
    int n    = gid >> 5;
    int lane = gid & 31;
    if (n >= N_NODES) return;
    int d0 = lane * 4;
    float4 acc = {0.f, 0.f, 0.f, 0.f};
    for (int f = 0; f < NAF; ++f) {
        int idx = x[n * NAF + f];
        float4 v = *(const float4*)(atom_emb + ((size_t)f * 64 + idx) * DIM + d0);
        acc.x += v.x; acc.y += v.y; acc.z += v.z; acc.w += v.w;
    }
    *(float4*)(h + (size_t)n * DIM + d0) = acc;
    _Float16* ph = h16 + (size_t)n * DIM + d0;
    ph[0] = (_Float16)acc.x; ph[1] = (_Float16)acc.y;
    ph[2] = (_Float16)acc.z; ph[3] = (_Float16)acc.w;
    if (lane == 0) deg[n] = 1.0f;
}

__global__ void k_deg_accum(const int* __restrict__ ei, float* __restrict__ deg)
{
    int e = blockIdx.x * blockDim.x + threadIdx.x;
    if (e >= N_EDGES) return;
    atomicAdd(&deg[ei[e]], 1.0f);   // row = source
}

__global__ void k_dis(const float* __restrict__ deg, float* __restrict__ dis)
{
    int n = blockIdx.x * blockDim.x + threadIdx.x;
    if (n >= N_NODES) return;
    dis[n] = rsqrtf(deg[n]);
}

__global__ void k_edge_norm(const int* __restrict__ ei, const float* __restrict__ dis,
                            float* __restrict__ nrm)
{
    int e = blockIdx.x * blockDim.x + threadIdx.x;
    if (e >= N_EDGES) return;
    nrm[e] = dis[ei[e]] * dis[ei[N_EDGES + e]];
}

__global__ void k_w_to_f16(const float* __restrict__ W, _Float16* __restrict__ w16)
{
    int i = blockIdx.x * blockDim.x + threadIdx.x;
    if (i >= NLAYERS * DIM * DIM) return;
    w16[i] = (_Float16)W[i];
}

__global__ void k_zero(float* __restrict__ p, int n4)
{
    int i = blockIdx.x * blockDim.x + threadIdx.x;
    if (i >= n4) return;
    float4 z = {0.f, 0.f, 0.f, 0.f};
    ((float4*)p)[i] = z;
}

// ---------------------------------------------------------------------------
// GEMM: xl[N,128] = h16[N,128] @ W16[128,128] + bias, via v_wmma_f32_16x16x32_f16
// Block = 256 threads = 8 waves; block computes 128 rows. A-tile (128x128 f16)
// staged via async global->LDS (no VGPR bounce) on full blocks; W^T-tile in LDS.
// Wave w owns row-tile w and sweeps 8 column tiles, 4 K-chunks each.
// Full blocks (781 of 782) take a guard-free store path with immediate offsets.
// ---------------------------------------------------------------------------
__global__ void __launch_bounds__(256) k_gemm_wmma(
    const _Float16* __restrict__ h16, const _Float16* __restrict__ w16l,
    const float* __restrict__ bias, float* __restrict__ xl)
{
    __shared__ _Float16 At[DIM * DIM];   // [row][k], row-major
    __shared__ _Float16 Bt[DIM * DIM];   // [n][k]  (W transposed)
    int t   = threadIdx.x;
    int r0b = blockIdx.x * DIM;
    const bool full = (r0b + DIM) <= N_NODES;     // block-uniform

    if (full) {
#if HAVE_ASYNC_LDS
        // Direct global -> LDS async copy, 16B per lane per issue.
        #pragma unroll
        for (int i = 0; i < 8; ++i) {
            int hoff = (t + i * 256) * 8;         // half offset in tile
            __builtin_amdgcn_global_load_async_to_lds_b128(
                (gv4i_p)(uintptr_t)(h16 + (size_t)r0b * DIM + hoff),
                (lv4i_p)(uint32_t)(uintptr_t)(At + hoff),
                0, 0);
        }
#else
        #pragma unroll
        for (int i = 0; i < 8; ++i) {
            int hoff = (t + i * 256) * 8;
            *(uint4*)(At + hoff) = *(const uint4*)(h16 + (size_t)r0b * DIM + hoff);
        }
#endif
    } else {
        for (int i = 0; i < 8; ++i) {
            int hoff = (t + i * 256) * 8;
            int row  = hoff >> 7;
            uint4 v  = {0u, 0u, 0u, 0u};
            if (r0b + row < N_NODES)
                v = *(const uint4*)(h16 + (size_t)(r0b + row) * DIM + (hoff & 127));
            *(uint4*)(At + hoff) = v;
        }
    }
    // Stage W transposed: each thread copies 64 contiguous halves of W[k][n]
    // into Bt[n][k].
    for (int i = 0; i < 64; ++i) {
        int idx = t * 64 + i;
        int k   = idx >> 7;
        int n   = idx & 127;
        Bt[n * DIM + k] = w16l[idx];
    }
#if HAVE_ASYNC_LDS
    if (full) wait_asynccnt0();
#endif
    __syncthreads();

    int wave = t >> 5, lane = t & 31;
    int m  = lane & 15;       // row/col within 16-tile
    int hl = lane >> 4;       // half-wave select (K +8 offset)

    union Frag { v16h v; uint4 u[2]; } fa[4], fb;
    int arow = wave * 16 + m;
    #pragma unroll
    for (int kc = 0; kc < 4; ++kc) {
        int kb = kc * 32 + hl * 8;
        fa[kc].u[0] = *(const uint4*)(At + arow * DIM + kb);
        fa[kc].u[1] = *(const uint4*)(At + arow * DIM + kb + 16);
    }

    #pragma unroll
    for (int ct = 0; ct < 8; ++ct) {
        v8f acc = {};
        int brow = ct * 16 + m;
        #pragma unroll
        for (int kc = 0; kc < 4; ++kc) {
            int kb = kc * 32 + hl * 8;
            fb.u[0] = *(const uint4*)(Bt + brow * DIM + kb);
            fb.u[1] = *(const uint4*)(Bt + brow * DIM + kb + 16);
            acc = __builtin_amdgcn_wmma_f32_16x16x32_f16(
                false, fa[kc].v, false, fb.v, (short)0, acc, false, false);
        }
        int c  = ct * 16 + m;
        float bv = bias[c];
        // C layout: VGPR j -> M = j + 8*hl, N = c
        if (full) {
            float* po = xl + (size_t)(r0b + wave * 16 + hl * 8) * DIM + c;
            #pragma unroll
            for (int j = 0; j < 8; ++j)
                po[(size_t)j * DIM] = acc[j] + bv;   // imm offsets j*512B
        } else {
            #pragma unroll
            for (int j = 0; j < 8; ++j) {
                int gr = r0b + wave * 16 + hl * 8 + j;
                if (gr < N_NODES) xl[(size_t)gr * DIM + c] = acc[j] + bv;
            }
        }
    }
}

// ---------------------------------------------------------------------------
// Fused edge kernel: e = sum bond-emb; msg = norm*relu(xl[row]+e);
// atomicAdd into agg[col]. One wave32 per edge, float4 per lane.
// ---------------------------------------------------------------------------
__global__ void __launch_bounds__(256) k_edge(
    const int* __restrict__ ei, const int* __restrict__ ea,
    const float* __restrict__ bondl, const float* __restrict__ xl,
    const float* __restrict__ nrm, float* __restrict__ agg)
{
    int gid = blockIdx.x * blockDim.x + threadIdx.x;
    int e   = gid >> 5;
    if (e >= N_EDGES) return;
    int lane = gid & 31, d0 = lane * 4;
    int row = ei[e], col = ei[N_EDGES + e];

    float4 s = {0.f, 0.f, 0.f, 0.f};
    #pragma unroll
    for (int f = 0; f < NBF; ++f) {
        int idx = ea[e * NBF + f];
        float4 v = *(const float4*)(bondl + ((size_t)f * BVOCAB + idx) * DIM + d0);
        s.x += v.x; s.y += v.y; s.z += v.z; s.w += v.w;
    }
    float4 xv = *(const float4*)(xl + (size_t)row * DIM + d0);
    float nm = nrm[e];
    float m0 = nm * fmaxf(xv.x + s.x, 0.f);
    float m1 = nm * fmaxf(xv.y + s.y, 0.f);
    float m2 = nm * fmaxf(xv.z + s.z, 0.f);
    float m3 = nm * fmaxf(xv.w + s.w, 0.f);
    float* pa = agg + (size_t)col * DIM + d0;
    atomicAdd(pa + 0, m0); atomicAdd(pa + 1, m1);
    atomicAdd(pa + 2, m2); atomicAdd(pa + 3, m3);
}

// ---------------------------------------------------------------------------
// Node update + LayerNorm (+optional relu). One wave32 per node.
// ---------------------------------------------------------------------------
__global__ void __launch_bounds__(256) k_node(
    const float* __restrict__ agg, const float* __restrict__ xl,
    const float* __restrict__ rootl, const float* __restrict__ gammal,
    const float* __restrict__ betal, const float* __restrict__ deg,
    float* __restrict__ hout, _Float16* __restrict__ h16out, int do_relu)
{
    int gid = blockIdx.x * blockDim.x + threadIdx.x;
    int n   = gid >> 5;
    if (n >= N_NODES) return;
    int lane = gid & 31, d0 = lane * 4;

    float invdeg = 1.0f / deg[n];
    float4 a  = *(const float4*)(agg + (size_t)n * DIM + d0);
    float4 xv = *(const float4*)(xl + (size_t)n * DIM + d0);
    float hn[4];
    hn[0] = a.x + fmaxf(xv.x + rootl[d0 + 0], 0.f) * invdeg;
    hn[1] = a.y + fmaxf(xv.y + rootl[d0 + 1], 0.f) * invdeg;
    hn[2] = a.z + fmaxf(xv.z + rootl[d0 + 2], 0.f) * invdeg;
    hn[3] = a.w + fmaxf(xv.w + rootl[d0 + 3], 0.f) * invdeg;

    float s = hn[0] + hn[1] + hn[2] + hn[3];
    for (int msk = 16; msk >= 1; msk >>= 1) s += __shfl_xor(s, msk, 32);
    float mu = s * (1.0f / DIM);

    float d0v = hn[0] - mu, d1v = hn[1] - mu, d2v = hn[2] - mu, d3v = hn[3] - mu;
    float s2 = d0v * d0v + d1v * d1v + d2v * d2v + d3v * d3v;
    for (int msk = 16; msk >= 1; msk >>= 1) s2 += __shfl_xor(s2, msk, 32);
    float inv = rsqrtf(s2 * (1.0f / DIM) + LN_EPS);

    float o[4];
    o[0] = d0v * inv * gammal[d0 + 0] + betal[d0 + 0];
    o[1] = d1v * inv * gammal[d0 + 1] + betal[d0 + 1];
    o[2] = d2v * inv * gammal[d0 + 2] + betal[d0 + 2];
    o[3] = d3v * inv * gammal[d0 + 3] + betal[d0 + 3];
    if (do_relu) {
        o[0] = fmaxf(o[0], 0.f); o[1] = fmaxf(o[1], 0.f);
        o[2] = fmaxf(o[2], 0.f); o[3] = fmaxf(o[3], 0.f);
    }
    float4 ov = {o[0], o[1], o[2], o[3]};
    *(float4*)(hout + (size_t)n * DIM + d0) = ov;
    if (h16out) {
        _Float16* ph = h16out + (size_t)n * DIM + d0;
        ph[0] = (_Float16)o[0]; ph[1] = (_Float16)o[1];
        ph[2] = (_Float16)o[2]; ph[3] = (_Float16)o[3];
    }
}

// ---------------------------------------------------------------------------
extern "C" void kernel_launch(void* const* d_in, const int* in_sizes, int n_in,
                              void* d_out, int out_size, void* d_ws, size_t ws_size,
                              hipStream_t stream)
{
    const int*   x        = (const int*)d_in[0];
    const int*   ei       = (const int*)d_in[1];
    const int*   ea       = (const int*)d_in[2];
    const float* atom_emb = (const float*)d_in[3];
    const float* W        = (const float*)d_in[4];
    const float* b        = (const float*)d_in[5];
    const float* root     = (const float*)d_in[6];
    const float* bond     = (const float*)d_in[7];
    const float* gamma    = (const float*)d_in[8];
    const float* beta     = (const float*)d_in[9];
    float* out = (float*)d_out;

    // Workspace carve-up (all 256B aligned)
    char* ws = (char*)d_ws;
    const size_t SZ_HF  = (size_t)N_NODES * DIM * 4;   // 51.2 MB
    const size_t SZ_H16 = (size_t)N_NODES * DIM * 2;   // 25.6 MB
    const size_t SZ_DEG = ((size_t)N_NODES * 4 + 255) & ~(size_t)255;
    const size_t SZ_NRM = (size_t)N_EDGES * 4;
    size_t off = 0;
    float*    hF  = (float*)   (ws + off); off += SZ_HF;
    _Float16* h16 = (_Float16*)(ws + off); off += SZ_H16;
    float*    xl  = (float*)   (ws + off); off += SZ_HF;
    float*    agg = (float*)   (ws + off); off += SZ_HF;
    float*    deg = (float*)   (ws + off); off += SZ_DEG;
    float*    dis = (float*)   (ws + off); off += SZ_DEG;
    float*    nrm = (float*)   (ws + off); off += ((SZ_NRM + 255) & ~(size_t)255);
    _Float16* w16 = (_Float16*)(ws + off);

    const int B = 256;
    int gNodeWave = (N_NODES * 32 + B - 1) / B;   // wave-per-node kernels
    int gEdgeWave = (N_EDGES * 32 + B - 1) / B;   // wave-per-edge kernels
    int gNode     = (N_NODES + B - 1) / B;
    int gEdge     = (N_EDGES + B - 1) / B;

    k_atom_encode<<<gNodeWave, B, 0, stream>>>(x, atom_emb, hF, h16, deg);
    k_deg_accum  <<<gEdge, B, 0, stream>>>(ei, deg);
    k_dis        <<<gNode, B, 0, stream>>>(deg, dis);
    k_edge_norm  <<<gEdge, B, 0, stream>>>(ei, dis, nrm);
    k_w_to_f16   <<<(NLAYERS * DIM * DIM + B - 1) / B, B, 0, stream>>>(W, w16);

    int n4 = N_NODES * DIM / 4;
    for (int l = 0; l < NLAYERS; ++l) {
        k_zero<<<(n4 + B - 1) / B, B, 0, stream>>>(agg, n4);
        k_gemm_wmma<<<(N_NODES + 127) / 128, B, 0, stream>>>(
            h16, w16 + (size_t)l * DIM * DIM, b + l * DIM, xl);
        k_edge<<<gEdgeWave, B, 0, stream>>>(
            ei, ea, bond + (size_t)l * NBF * BVOCAB * DIM, xl, nrm, agg);
        bool last = (l == NLAYERS - 1);
        k_node<<<gNodeWave, B, 0, stream>>>(
            agg, xl, root + l * DIM, gamma + l * DIM, beta + l * DIM, deg,
            last ? out : hF, last ? (_Float16*)nullptr : h16, last ? 0 : 1);
    }
}
